// SimpleSSC_VanillaSFNN_86474871538188
// MI455X (gfx1250) — matrix operand order repaired and checked by reading, not compiled, not run on hardware
//
#include <hip/hip_runtime.h>
#include <hip/hip_bf16.h>
#include <math.h>

typedef __attribute__((ext_vector_type(16))) __bf16         bf16x16;
typedef __attribute__((ext_vector_type(8)))  float          v8f;
typedef __attribute__((ext_vector_type(4)))  unsigned int   u32x4;
typedef __attribute__((ext_vector_type(4)))  float          f32x4;
typedef __attribute__((ext_vector_type(4)))  unsigned short u16x4;

union FragU { u32x4 q[2]; bf16x16 v; };

__device__ __forceinline__ unsigned short f2bf(float f) {
  unsigned int u = __float_as_uint(f);
  u += 0x7FFFu + ((u >> 16) & 1u);           // round-to-nearest-even
  return (unsigned short)(u >> 16);
}
__device__ __forceinline__ float bf2f(unsigned short s) {
  return __uint_as_float(((unsigned int)s) << 16);
}

// 16-bit A matrix 16x32 (MxK): lane<16 holds M=lane, K=0..7 (+0B) and K=16..23 (+32B);
// lane>=16 holds M=lane-16, K=8..15 and K=24..31.  rs = LDS row stride in ushorts.
__device__ __forceinline__ bf16x16 frag_a(const unsigned short* base, int rs, int lane) {
  int m = lane & 15, hi = lane >> 4;
  const unsigned short* p = base + m * rs + hi * 8;
  FragU f;
  f.q[0] = *(const u32x4*)(p);
  f.q[1] = *(const u32x4*)(p + 16);
  return f.v;
}
// 16-bit B matrix 32x16 (KxN), stored n-major in LDS (Bt[n][k]): lane<16 holds N=lane,
// K=0..7 and K=16..23; lane>=16 holds N=lane-16, K=8..15 and K=24..31.
__device__ __forceinline__ bf16x16 frag_b(const unsigned short* base, int rs, int lane) {
  int n = lane & 15, hi = lane >> 4;
  const unsigned short* p = base + n * rs + hi * 8;
  FragU f;
  f.q[0] = *(const u32x4*)(p);
  f.q[1] = *(const u32x4*)(p + 16);
  return f.v;
}

// ---------------- Kernel A: d_all[t,b,h] = x[b,t,:] @ W_dense + b_dense (bf16 out) ----
#define TM 128
#define TN 128
#define TK 32
#define SA 40   // LDS row stride (ushorts), 80B: keeps 16B alignment, spreads banks
#define SBK 40

__global__ __launch_bounds__(256) void k_in_gemm(
    const float* __restrict__ x, const float* __restrict__ Wd,
    const float* __restrict__ bd, unsigned short* __restrict__ dall)
{
  __shared__ __attribute__((aligned(16))) unsigned short aLds[TM * SA];
  __shared__ __attribute__((aligned(16))) unsigned short bLds[TN * SBK];
  const int tid = threadIdx.x;
  const int lane = tid & 31, w = tid >> 5;
  const int wm = w & 3, wn = w >> 2;                 // 4x2 wave grid
  const int m0 = blockIdx.x * TM;                    // rows = b*T + t
  const int n0 = blockIdx.y * TN;                    // cols = h

  v8f c[2][4];
#pragma unroll
  for (int mt = 0; mt < 2; ++mt)
#pragma unroll
    for (int nt = 0; nt < 4; ++nt) c[mt][nt] = 0.0f;

  for (int k0 = 0; k0 < 700; k0 += TK) {
    __syncthreads();
    if (k0 + TK <= 700) {
      // full tile: guard-free, vectorized staging
      // A tile 128x32: float4 loads (x rows are 2800B = 16B aligned), u16x4 stores
#pragma unroll
      for (int e = tid; e < TM * TK / 4; e += 256) {   // 4 iters
        int r = e >> 3, k4 = (e & 7) * 4;
        f32x4 v = *(const f32x4*)(x + (size_t)(m0 + r) * 700 + k0 + k4);
        u16x4 o = { f2bf(v.x), f2bf(v.y), f2bf(v.z), f2bf(v.w) };
        *(u16x4*)(aLds + r * SA + k4) = o;
      }
      // B tile 32x128 -> transposed [n][k]: float4 loads over n, 4 b16 stores
#pragma unroll
      for (int e = tid; e < TK * TN / 4; e += 256) {   // 4 iters
        int k = e >> 5, n4 = (e & 31) * 4;
        f32x4 v = *(const f32x4*)(Wd + (size_t)(k0 + k) * 1024 + n0 + n4);
        bLds[(n4 + 0) * SBK + k] = f2bf(v.x);
        bLds[(n4 + 1) * SBK + k] = f2bf(v.y);
        bLds[(n4 + 2) * SBK + k] = f2bf(v.z);
        bLds[(n4 + 3) * SBK + k] = f2bf(v.w);
      }
    } else {
      // tail tile (k0 = 672, 28 valid K): guarded scalar staging
      for (int e = tid; e < TM * TK; e += 256) {
        int r = e >> 5, k = e & 31;
        int kk = k0 + k;
        float v = (kk < 700) ? x[(size_t)(m0 + r) * 700 + kk] : 0.0f;
        aLds[r * SA + k] = f2bf(v);
      }
      for (int e = tid; e < TK * TN; e += 256) {
        int k = e >> 7, n = e & 127;
        int kk = k0 + k;
        float v = (kk < 700) ? Wd[(size_t)kk * 1024 + n0 + n] : 0.0f;
        bLds[n * SBK + k] = f2bf(v);
      }
    }
    __syncthreads();

    bf16x16 af[2];
#pragma unroll
    for (int mt = 0; mt < 2; ++mt)
      af[mt] = frag_a(aLds + (wm * 32 + mt * 16) * SA, SA, lane);
#pragma unroll
    for (int nt = 0; nt < 4; ++nt) {
      bf16x16 bfr = frag_b(bLds + (wn * 64 + nt * 16) * SBK, SBK, lane);
#pragma unroll
      for (int mt = 0; mt < 2; ++mt)
        c[mt][nt] = __builtin_amdgcn_wmma_f32_16x16x32_bf16(
            false, af[mt], false, bfr, (short)0, c[mt][nt], false, false);
    }
  }

  // epilogue: +bias, remap row (b*T+t) -> [t,b,h], store bf16
  const int hi = lane >> 4, nl = lane & 15;
#pragma unroll
  for (int mt = 0; mt < 2; ++mt) {
#pragma unroll
    for (int nt = 0; nt < 4; ++nt) {
      int n = n0 + wn * 64 + nt * 16 + nl;
      float bias = bd[n];
#pragma unroll
      for (int r = 0; r < 8; ++r) {
        int m = m0 + wm * 32 + mt * 16 + hi * 8 + r;
        int b = m / 250, t = m - b * 250;
        dall[((size_t)(t * 256 + b) << 10) + n] = f2bf(c[mt][nt][r] + bias);
      }
    }
  }
}

// ---------------- Kernel B: LIF recurrence, thread per (b,h); spikes in place ---------
__global__ __launch_bounds__(256) void k_scan(
    const float* __restrict__ tau, const float* __restrict__ mem0,
    const float* __restrict__ sp0, unsigned short* __restrict__ dall)
{
  int gid = blockIdx.x * 256 + threadIdx.x;         // 0 .. B*H-1
  int b = gid >> 10, h = gid & 1023;
  float alpha = 1.0f / (1.0f + __expf(-tau[h]));
  float one_m = 1.0f - alpha;
  float mem   = mem0[gid];
  float spike = sp0[gid];
  size_t idx = ((size_t)b << 10) + h;
  const size_t STRIDE = (size_t)256 * 1024;
  for (int t = 0; t < 250; ++t, idx += STRIDE) {
    float d = bf2f(dall[idx]);
    if (t < 249) __builtin_prefetch(&dall[idx + STRIDE], 0, 1);   // global_prefetch_b8
    mem = mem * alpha + one_m * d - spike;          // VTH = 1
    spike = (mem - 1.0f > 0.0f) ? 1.0f : 0.0f;
    dall[idx] = spike != 0.0f ? (unsigned short)0x3F80u : (unsigned short)0;
  }
}

// ---------------- Kernel C: logits = spike @ W_out + b_out, softmax, masked acc -------
#define CSA 40
__global__ __launch_bounds__(256) void k_out(
    const unsigned short* __restrict__ dall, const float* __restrict__ Wo,
    const float* __restrict__ bo, float* __restrict__ partial)
{
  __shared__ __attribute__((aligned(16))) unsigned short wLds[48 * 1032]; // W_out^T bf16
  __shared__ __attribute__((aligned(16))) unsigned short aLds[8][16 * CSA];
  __shared__ float lgW[8][16 * 48];
  __shared__ float accW[8][16 * 36];

  const int tid = threadIdx.x, lane = tid & 31, w = tid >> 5;
  const int btile = blockIdx.x;                     // 0..15 (16 batches each)
  const int chunk = blockIdx.y;                     // 0..7  (32 timesteps each)

  for (int e = tid; e < 48 * 1024; e += 256) {      // stage W_out -> bf16, [n][k]
    int k = e / 48, n = e - k * 48;
    float v = (n < 35) ? Wo[k * 35 + n] : 0.0f;     // pad O 35->48 with zeros
    wLds[n * 1032 + k] = f2bf(v);
  }
  for (int e = tid; e < 8 * 16 * 36; e += 256) ((float*)accW)[e] = 0.0f;
  __syncthreads();

  const int hi = lane >> 4, nl = lane & 15;
  for (int i = 0; i < 4; ++i) {
    int t = chunk * 32 + w * 4 + i;                 // each wave owns 4 timesteps
    if (t < 250) {
      v8f c[3]; c[0] = 0.0f; c[1] = 0.0f; c[2] = 0.0f;
      const size_t rowbase = ((size_t)(t * 256 + btile * 16) << 10);
      unsigned short* am = aLds[w];
      const int r = lane >> 1, ks = (lane & 1) * 16;
      unsigned short* ldst = am + r * CSA + ks;               // 16B aligned
      const unsigned int lds_off = (unsigned int)(size_t)ldst; // AS(3) offset bits
      for (int kb = 0; kb < 32; ++kb) {
        // wave-private A stage: 16 batches x 32 k of bf16 spikes via CDNA5
        // async global->LDS DMA (INST_OFFSET applies to both LDS and mem addr)
        const unsigned short* gp = dall + rowbase + ((size_t)r << 10) + kb * 32 + ks;
        asm volatile("global_load_async_to_lds_b128 %0, %1, off\n\t"
                     "global_load_async_to_lds_b128 %0, %1, off offset:16"
                     :: "v"(lds_off), "v"(gp)
                     : "memory");
        asm volatile("s_wait_asynccnt 0x0" ::: "memory");
        bf16x16 a = frag_a(am, CSA, lane);
#pragma unroll
        for (int nt = 0; nt < 3; ++nt) {
          bf16x16 bfr = frag_b(wLds + nt * 16 * 1032 + kb * 32, 1032, lane);
          c[nt] = __builtin_amdgcn_wmma_f32_16x16x32_bf16(
              false, a, false, bfr, (short)0, c[nt], false, false);
        }
      }
      // dump logits to LDS: C layout vgpr r -> M = r (+8 for high lanes), N = lane&15
#pragma unroll
      for (int nt = 0; nt < 3; ++nt)
#pragma unroll
        for (int rr = 0; rr < 8; ++rr)
          lgW[w][(hi * 8 + rr) * 48 + nt * 16 + nl] = c[nt][rr];
      if (t > 10 && lane < 16) {                    // softmax per batch row, masked acc
        float* lg = &lgW[w][lane * 48];
        float mx = -3.4e38f;
        for (int o = 0; o < 35; ++o) { float v = lg[o] + bo[o]; lg[o] = v; mx = fmaxf(mx, v); }
        float s = 0.0f;
        for (int o = 0; o < 35; ++o) { float e = __expf(lg[o] - mx); lg[o] = e; s += e; }
        float inv = 1.0f / s;
        float* ac = &accW[w][lane * 36];
        for (int o = 0; o < 35; ++o) ac[o] += lg[o] * inv;
      }
    }
  }
  __syncthreads();
  // fixed-order cross-wave reduce -> partial[chunk][b][o]   (deterministic)
  for (int e = tid; e < 16 * 35; e += 256) {
    int r = e / 35, o = e - r * 35;
    float s = 0.0f;
#pragma unroll
    for (int ww = 0; ww < 8; ++ww) s += accW[ww][r * 36 + o];
    partial[(((size_t)chunk * 256) + btile * 16 + r) * 35 + o] = s;
  }
}

// ---------------- Kernel D: fixed-order chunk reduction -> d_out ----------------------
__global__ __launch_bounds__(256) void k_reduce(
    const float* __restrict__ partial, float* __restrict__ out)
{
  int e = blockIdx.x * 256 + threadIdx.x;
  if (e >= 256 * 35) return;
  float s = 0.0f;
#pragma unroll
  for (int ch = 0; ch < 8; ++ch) s += partial[(size_t)ch * 256 * 35 + e];
  out[e] = s;
}

extern "C" void kernel_launch(void* const* d_in, const int* in_sizes, int n_in,
                              void* d_out, int out_size, void* d_ws, size_t ws_size,
                              hipStream_t stream) {
  const float* x   = (const float*)d_in[0];   // [256,250,700]
  const float* Wd  = (const float*)d_in[1];   // [700,1024]
  const float* bd  = (const float*)d_in[2];   // [1024]
  const float* Wo  = (const float*)d_in[3];   // [1024,35]
  const float* bo  = (const float*)d_in[4];   // [35]
  const float* tau = (const float*)d_in[5];   // [1024]
  const float* m0  = (const float*)d_in[6];   // [256,1024]
  const float* s0  = (const float*)d_in[7];   // [256,1024]

  unsigned short* dall = (unsigned short*)d_ws;                          // bf16 [250,256,1024] = 131 MB (fits L2)
  float* partial = (float*)((char*)d_ws + (size_t)250 * 256 * 1024 * 2); // [8,256,35]

  hipLaunchKernelGGL(k_in_gemm, dim3(500, 8), dim3(256), 0, stream, x, Wd, bd, dall);
  hipLaunchKernelGGL(k_scan,    dim3(1024),   dim3(256), 0, stream, tau, m0, s0, dall);
  hipLaunchKernelGGL(k_out,     dim3(16, 8),  dim3(256), 0, stream, dall, Wo, bo, partial);
  hipLaunchKernelGGL(k_reduce,  dim3(35),     dim3(256), 0, stream, partial, (float*)d_out);
}
